// pos_lstm_model_68556267979012
// MI455X (gfx1250) — matrix operand order repaired and checked by reading, not compiled, not run on hardware
//
#include <hip/hip_runtime.h>
#include <hip/hip_bf16.h>
#include <cstdint>

typedef __attribute__((ext_vector_type(16))) __bf16 v16bf;
typedef __attribute__((ext_vector_type(8)))  __bf16 v8bf;
typedef __attribute__((ext_vector_type(8)))  float  v8f;

#define N_WORDS 8192
#define T_CHARS 16
#define CDIM    128
#define WDIM    300
#define HDIM    512
#define PDIM    64
#define NCHARS  256
#define KIN     428          // WD + C
#define KX      448          // KIN padded to multiple of 32
#define G4C     (4*CDIM)     // 512
#define G4H     (4*HDIM)     // 2048

__device__ __forceinline__ float sigm(float x){ return 1.0f/(1.0f+__expf(-x)); }

// ---- WMMA fragment loaders (16x16x32 bf16; contiguous k = 16*half + e in A and B, not the ISA table order) ----
// A (16x32): lanes 0-15 -> M=lane, k-half 0; lanes 16-31 -> M=lane-16, k-half 1.
// elements e=0..15 -> K = khalf*16 + e (matches B fragment).
__device__ __forceinline__ v16bf load_a_frag(const __bf16* A, int lda, int k0)
{
  const int lane = threadIdx.x & 31;
  const int m    = lane & 15;
  const int kh   = (lane >> 4) * 16;
  const __bf16* p0 = A + m*lda + k0 + kh;
  v8bf lo = *(const v8bf*)p0;          // K = k0+kh .. +8
  v8bf hi = *(const v8bf*)(p0 + 8);    // K = k0+kh+8 .. +8
  v16bf r;
  #pragma unroll
  for (int i = 0; i < 8; ++i){ r[i] = lo[i]; r[i+8] = hi[i]; }
  return r;
}
// B (32x16): col n = lane&15; lanes 0-15 hold K=0..15, lanes 16-31 K=16..31.
// W is row-major [out_cols][k] (i.e. the PyTorch weight layout, gates = x @ W^T).
__device__ __forceinline__ v16bf load_b_frag(const __bf16* W, int ldb, int nb, int k0)
{
  const int lane = threadIdx.x & 31;
  const int n    = lane & 15;
  const int kh   = (lane >> 4) * 16;
  return *(const v16bf*)(W + (size_t)(nb + n)*ldb + k0 + kh);
}

__device__ __forceinline__ v8f wmma_bf16(v16bf a, v16bf b, v8f c){
  return __builtin_amdgcn_wmma_f32_16x16x32_bf16(false, a, false, b, (short)0, c, false, false);
}

// ---------------- prep: f32 -> bf16 weight/activation staging ----------------
__global__ void k_prep(const float* cWih, const float* cWhh, const float* cemb,
                       const float* fcW, const float* wWih, const float* wemb,
                       __bf16* cWih_b, __bf16* cWhh_b, __bf16* cemb_b,
                       __bf16* fcW_b, __bf16* wWih_b, __bf16* xin_b)
{
  const long S0 = (long)G4C*CDIM;          // 65536
  const long S1 = (long)G4C*CDIM;
  const long S2 = (long)NCHARS*CDIM;       // 32768
  const long S3 = (long)PDIM*HDIM;         // 32768
  const long S4 = (long)G4H*KX;            // 917504
  const long S5 = (long)N_WORDS*KX;        // 3670016
  const long total = S0+S1+S2+S3+S4+S5;
  for (long i = blockIdx.x*(long)blockDim.x + threadIdx.x; i < total;
       i += (long)gridDim.x*blockDim.x){
    long j = i;
    if (j < S0){ cWih_b[j] = (__bf16)cWih[j]; continue; } j -= S0;
    if (j < S1){ cWhh_b[j] = (__bf16)cWhh[j]; continue; } j -= S1;
    if (j < S2){ cemb_b[j] = (__bf16)cemb[j]; continue; } j -= S2;
    if (j < S3){ fcW_b[j]  = (__bf16)fcW[j];  continue; } j -= S3;
    if (j < S4){
      long r = j / KX, k = j % KX;
      wWih_b[j] = (k < KIN) ? (__bf16)wWih[r*KIN + k] : (__bf16)0.0f;
      continue;
    } j -= S4;
    { long w = j / KX, k = j % KX;
      if (k < WDIM)      xin_b[j] = (__bf16)wemb[w*WDIM + k];
      else if (k >= KIN) xin_b[j] = (__bf16)0.0f;
      // cols [WDIM, KIN) = char_last, filled by k_char_lstm
    }
  }
}

// ---------------- phase A: batched char LSTM (WMMA) ----------------
// block = 16 words, 8 waves; wave wv owns hidden cols [wv*16, wv*16+16) of each gate
__global__ void k_char_lstm(const int* char_idx, const int* lengths,
                            const __bf16* cemb_b, const __bf16* cWih_b,
                            const __bf16* cWhh_b, const float* c_b, __bf16* xin_b)
{
  __shared__ __bf16 xs[16*CDIM];
  __shared__ __bf16 hsb[16*CDIM];
  __shared__ int lens[16];

  const int tid   = threadIdx.x;
  const int wv    = tid >> 5;
  const int lane  = tid & 31;
  const int wbase = blockIdx.x * 16;
  const int ncol  = lane & 15;
  const int mbase = (lane >> 4) * 8;
  const int colbase = wv * 16;

  if (tid < 16) lens[tid] = lengths[wbase + tid];
  for (int i = tid; i < 16*CDIM; i += 256) hsb[i] = (__bf16)0.0f;

  const float bi = c_b[0*CDIM + colbase + ncol];
  const float bf = c_b[1*CDIM + colbase + ncol];
  const float bg = c_b[2*CDIM + colbase + ncol];
  const float bo = c_b[3*CDIM + colbase + ncol];
  v8f cst = {};
  __syncthreads();

  for (int t = 0; t < T_CHARS; ++t){
    { // gather x_t = char_emb[char_idx[:, t]] into LDS (8 bf16 / thread)
      const int base = tid * 8;
      const int m  = base >> 7;
      const int c0 = base & 127;
      const int ci = char_idx[(wbase + m)*T_CHARS + t];
      *(uint4*)&xs[m*CDIM + c0] = *(const uint4*)&cemb_b[ci*CDIM + c0];
    }
    __syncthreads();

    v16bf ax[4], ah[4];
    #pragma unroll
    for (int kt = 0; kt < 4; ++kt){
      ax[kt] = load_a_frag(xs,  CDIM, kt*32);
      ah[kt] = load_a_frag(hsb, CDIM, kt*32);
    }
    v8f acc[4];
    #pragma unroll
    for (int q = 0; q < 4; ++q){
      v8f a = {};
      #pragma unroll
      for (int kt = 0; kt < 4; ++kt)
        a = wmma_bf16(ax[kt], load_b_frag(cWih_b, CDIM, q*CDIM + colbase, kt*32), a);
      #pragma unroll
      for (int kt = 0; kt < 4; ++kt)
        a = wmma_bf16(ah[kt], load_b_frag(cWhh_b, CDIM, q*CDIM + colbase, kt*32), a);
      acc[q] = a;
    }
    __syncthreads();   // all waves done reading hsb -> safe to overwrite

    #pragma unroll
    for (int r = 0; r < 8; ++r){
      const int m = mbase + r;
      const float iv = sigm(acc[0][r] + bi);
      const float fv = sigm(acc[1][r] + bf);
      const float gv = tanhf(acc[2][r] + bg);
      const float ov = sigm(acc[3][r] + bo);
      const float cv = fv * cst[r] + iv * gv;
      cst[r] = cv;
      const float hv = ov * tanhf(cv);
      hsb[m*CDIM + colbase + ncol] = (__bf16)hv;
      if (t == lens[m] - 1)   // last valid step -> char_last into xin
        xin_b[(size_t)(wbase + m)*KX + WDIM + colbase + ncol] = (__bf16)hv;
    }
  }
}

// ---------------- phase B: gi = xin @ w_Wih^T + w_b  (WMMA GEMM) ----------------
__global__ void k_gemm_gi(const __bf16* xin_b, const __bf16* wWih_b,
                          const float* w_b, __bf16* gi_b)
{
  __shared__ __bf16 xt[16*KX];
  const int tid  = threadIdx.x;
  const int wv   = tid >> 5;
  const int lane = tid & 31;
  const int row0 = blockIdx.x * 16;
  const int nb   = blockIdx.y * 128 + wv * 16;

  for (int c = tid; c < 16*KX/8; c += 256)
    *(uint4*)&xt[c*8] = *(const uint4*)&xin_b[(size_t)row0*KX + c*8];
  __syncthreads();

  const float bias = w_b[nb + (lane & 15)];
  v8f acc = {};
  #pragma unroll
  for (int kt = 0; kt < KX/32; ++kt)
    acc = wmma_bf16(load_a_frag(xt, KX, kt*32), load_b_frag(wWih_b, KX, nb, kt*32), acc);

  const int n  = lane & 15;
  const int mb = (lane >> 4) * 8;
  #pragma unroll
  for (int r = 0; r < 8; ++r)
    gi_b[(size_t)(row0 + mb + r)*G4H + nb + n] = (__bf16)(acc[r] + bias);
}

// ---------------- phase C init + sequential word LSTM ----------------
__global__ void k_init(float* hbuf, unsigned* bar){
  if (threadIdx.x < HDIM) hbuf[threadIdx.x] = 0.0f;
  if (threadIdx.x == 0)   *bar = 0u;
}

// 16 persistent blocks; block owns hidden units [bid*32, bid*32+32) -> 128 gate rows.
// Weights register-resident (128 f32/thread); h broadcast via L2 + atomic barrier.
__global__ void __launch_bounds__(512, 1)
k_word_lstm(const __bf16* gi_b, const float* wWhh, float* hbuf,
            unsigned* bar, __bf16* outs_b)
{
  __shared__ float h_l[HDIM];
  __shared__ float part_s[128][4];
  __shared__ float gates[128];
  __shared__ float cstate[32];

  const int tid    = threadIdx.x;
  const int row_id = tid >> 2;            // 0..127
  const int part   = tid & 3;             // k-quarter
  const int q      = row_id >> 5;         // gate (i,f,g,o)
  const int jl     = row_id & 31;         // local hidden unit
  const int grow   = q*HDIM + blockIdx.x*32 + jl;

  float wreg[128];
  {
    const float* wr = wWhh + (size_t)grow*HDIM + part*128;
    #pragma unroll
    for (int i = 0; i < 128; i += 4){
      float4 v = *(const float4*)(wr + i);
      wreg[i] = v.x; wreg[i+1] = v.y; wreg[i+2] = v.z; wreg[i+3] = v.w;
    }
  }
  h_l[tid] = 0.0f;
  if (tid < 32) cstate[tid] = 0.0f;
  __syncthreads();

  for (int n = 0; n < N_WORDS; ++n){
    if (tid < 128 && n + 1 < N_WORDS)        // gfx1250 global_prefetch_b8
      __builtin_prefetch(gi_b + (size_t)(n+1)*G4H + grow, 0, 0);

    float p = 0.f;
    #pragma unroll
    for (int i = 0; i < 128; i += 4){
      float4 hv = *(const float4*)&h_l[part*128 + i];
      p += wreg[i]*hv.x + wreg[i+1]*hv.y + wreg[i+2]*hv.z + wreg[i+3]*hv.w;
    }
    part_s[row_id][part] = p;
    __syncthreads();

    if (tid < 128){
      const float s = part_s[tid][0] + part_s[tid][1] + part_s[tid][2] + part_s[tid][3];
      const int gq = tid >> 5, gj = tid & 31;
      gates[tid] = s + (float)gi_b[(size_t)n*G4H + gq*HDIM + blockIdx.x*32 + gj];
    }
    __syncthreads();

    if (tid < 32){
      const float iv = sigm(gates[tid]);
      const float fv = sigm(gates[32 + tid]);
      const float gv = tanhf(gates[64 + tid]);
      const float ov = sigm(gates[96 + tid]);
      const float cv = fv * cstate[tid] + iv * gv;
      cstate[tid] = cv;
      const float hv = ov * tanhf(cv);
      __hip_atomic_store(&hbuf[blockIdx.x*32 + tid], hv,
                         __ATOMIC_RELEASE, __HIP_MEMORY_SCOPE_AGENT);
      outs_b[(size_t)n*HDIM + blockIdx.x*32 + tid] = (__bf16)hv;
    }
    __syncthreads();

    if (tid == 0){   // monotonic 16-block barrier
      __hip_atomic_fetch_add(bar, 1u, __ATOMIC_ACQ_REL, __HIP_MEMORY_SCOPE_AGENT);
      const unsigned target = 16u * (unsigned)(n + 1);
      while (__hip_atomic_load(bar, __ATOMIC_ACQUIRE, __HIP_MEMORY_SCOPE_AGENT) < target)
        __builtin_amdgcn_s_sleep(1);
    }
    __syncthreads();

    h_l[tid] = __hip_atomic_load(&hbuf[tid], __ATOMIC_ACQUIRE, __HIP_MEMORY_SCOPE_AGENT);
    __syncthreads();
  }
}

// ---------------- phase D: logits = outs @ fc_W^T + fc_b, log_softmax ----------------
__global__ void k_fc_softmax(const __bf16* outs_b, const __bf16* fcW_b,
                             const float* fc_b, float* out)
{
  __shared__ __bf16 ot[16*HDIM];
  __shared__ float  lg[16*PDIM];
  const int tid  = threadIdx.x;    // 128 threads = 4 waves
  const int wv   = tid >> 5;
  const int lane = tid & 31;
  const int row0 = blockIdx.x * 16;

  for (int c = tid; c < 16*HDIM/8; c += 128)
    *(uint4*)&ot[c*8] = *(const uint4*)&outs_b[(size_t)row0*HDIM + c*8];
  __syncthreads();

  const int nb = wv * 16;
  const float bias = fc_b[nb + (lane & 15)];
  v8f acc = {};
  #pragma unroll
  for (int kt = 0; kt < HDIM/32; ++kt)
    acc = wmma_bf16(load_a_frag(ot, HDIM, kt*32), load_b_frag(fcW_b, HDIM, nb, kt*32), acc);

  const int n  = lane & 15;
  const int mb = (lane >> 4) * 8;
  #pragma unroll
  for (int r = 0; r < 8; ++r)
    lg[(mb + r)*PDIM + nb + n] = acc[r] + bias;
  __syncthreads();

  if (tid < 16){
    float mx = -1e30f;
    for (int c = 0; c < PDIM; ++c) mx = fmaxf(mx, lg[tid*PDIM + c]);
    float s = 0.f;
    for (int c = 0; c < PDIM; ++c) s += __expf(lg[tid*PDIM + c] - mx);
    const float l = __logf(s) + mx;
    for (int c = 0; c < PDIM; ++c)
      out[(size_t)(row0 + tid)*PDIM + c] = lg[tid*PDIM + c] - l;
  }
}

// ---------------- host launcher ----------------
extern "C" void kernel_launch(void* const* d_in, const int* in_sizes, int n_in,
                              void* d_out, int out_size, void* d_ws, size_t ws_size,
                              hipStream_t stream)
{
  const int*   char_idx = (const int*)  d_in[0];
  const int*   lengths  = (const int*)  d_in[1];
  const float* wemb     = (const float*)d_in[2];
  const float* cemb     = (const float*)d_in[3];
  const float* cWih     = (const float*)d_in[4];
  const float* cWhh     = (const float*)d_in[5];
  const float* c_b      = (const float*)d_in[6];
  const float* wWih     = (const float*)d_in[7];
  const float* wWhh     = (const float*)d_in[8];
  const float* w_b      = (const float*)d_in[9];
  const float* fcW      = (const float*)d_in[10];
  const float* fc_b     = (const float*)d_in[11];
  float* out = (float*)d_out;

  char* ws = (char*)d_ws;
  __bf16*   cWih_b = (__bf16*)(ws + 0);          //  131072 B
  __bf16*   cWhh_b = (__bf16*)(ws + 131072);     //  131072 B
  __bf16*   cemb_b = (__bf16*)(ws + 262144);     //   65536 B
  __bf16*   fcW_b  = (__bf16*)(ws + 327680);     //   65536 B
  __bf16*   wWih_b = (__bf16*)(ws + 393216);     // 1835008 B  [2048][448]
  __bf16*   xin_b  = (__bf16*)(ws + 2228224);    // 7340032 B  [8192][448]
  __bf16*   gi_b   = (__bf16*)(ws + 9568256);    // 33554432 B [8192][2048]
  __bf16*   outs_b = (__bf16*)(ws + 43122688);   // 8388608 B  [8192][512]
  float*    hbuf   = (float*) (ws + 51511296);   // 2048 B
  unsigned* bar    = (unsigned*)(ws + 51513344); // 4 B

  k_prep      <<<2048, 256, 0, stream>>>(cWih, cWhh, cemb, fcW, wWih, wemb,
                                         cWih_b, cWhh_b, cemb_b, fcW_b, wWih_b, xin_b);
  k_char_lstm <<<N_WORDS/16, 256, 0, stream>>>(char_idx, lengths, cemb_b,
                                               cWih_b, cWhh_b, c_b, xin_b);
  k_gemm_gi   <<<dim3(N_WORDS/16, G4H/128), 256, 0, stream>>>(xin_b, wWih_b, w_b, gi_b);
  k_init      <<<1, 512, 0, stream>>>(hbuf, bar);
  k_word_lstm <<<16, 512, 0, stream>>>(gi_b, wWhh, hbuf, bar, outs_b);
  k_fc_softmax<<<N_WORDS/16, 128, 0, stream>>>(outs_b, fcW_b, fc_b, out);
}
